// TransformerEncoderLayer_88356067213603
// MI455X (gfx1250) — matrix operand order repaired and checked by reading, not compiled, not run on hardware
//
#include <hip/hip_runtime.h>
#include <cstdint>
#include <cstddef>

// ---------------- CDNA5 WMMA types ----------------
typedef __attribute__((ext_vector_type(16))) _Float16 v16h;
typedef __attribute__((ext_vector_type(8)))  float    v8f;

union V16 { v16h v; float4 f[2]; };

#define TOK   25216          // 128 * 197 tokens
#define CDIM  768
#define MDIM  3072
#define NHEAD 12
#define HD    64
#define LSEQ  197

// CDNA5 async global->LDS copy (ASYNCcnt path), guarded so the file still
// compiles on host pass / toolchains without the builtin.
// Signature (from clang diagnostic): (int4 AS1*, int4 AS3*, imm offset, imm cpol)
#if defined(__AMDGCN__) && __has_builtin(__builtin_amdgcn_global_load_async_to_lds_b128)
#define USE_ASYNC_LDS 1
typedef int i32x4_t __attribute__((vector_size(16)));
typedef __attribute__((address_space(1))) i32x4_t gas_i4;
typedef __attribute__((address_space(3))) i32x4_t las_i4;
#else
#define USE_ASYNC_LDS 0
#endif

__device__ __forceinline__ void wait_asynccnt0() {
#if defined(__AMDGCN__)
#if __has_builtin(__builtin_amdgcn_s_wait_asynccnt)
  __builtin_amdgcn_s_wait_asynccnt(0);
#else
  asm volatile("s_wait_asynccnt 0x0" ::: "memory");
#endif
#endif
}

// ---------------- weight convert f32[K][N] -> f16[N][K] (transposed) ----------------
__global__ __launch_bounds__(256)
void cvtT_f32_f16_kernel(const float* __restrict__ s, _Float16* __restrict__ d, int K, int N) {
  int i = blockIdx.x * 256 + threadIdx.x;
  if (i < K * N) {
    int n = i / K, k = i - n * K;
    d[i] = (_Float16)s[(size_t)k * N + n];   // coalesced writes
  }
}

// ---------------- LayerNorm (768 ch), f32 in -> f16 out ----------------
__global__ __launch_bounds__(256)
void layernorm_f16_kernel(const float* __restrict__ x, const float* __restrict__ g,
                          const float* __restrict__ b, _Float16* __restrict__ out) {
  __shared__ float red[256];
  int t = blockIdx.x;
  int tid = threadIdx.x;
  const float* xr = x + (size_t)t * CDIM;
  float v0 = xr[tid], v1 = xr[tid + 256], v2 = xr[tid + 512];
  red[tid] = v0 + v1 + v2;
  __syncthreads();
  for (int o = 128; o > 0; o >>= 1) { if (tid < o) red[tid] += red[tid + o]; __syncthreads(); }
  float mu = red[0] * (1.f / 768.f);
  __syncthreads();
  float d0 = v0 - mu, d1 = v1 - mu, d2 = v2 - mu;
  red[tid] = d0 * d0 + d1 * d1 + d2 * d2;
  __syncthreads();
  for (int o = 128; o > 0; o >>= 1) { if (tid < o) red[tid] += red[tid + o]; __syncthreads(); }
  float rstd = rsqrtf(red[0] * (1.f / 768.f) + 1e-5f);
  _Float16* op = out + (size_t)t * CDIM;
  op[tid]       = (_Float16)(d0 * rstd * g[tid]       + b[tid]);
  op[tid + 256] = (_Float16)(d1 * rstd * g[tid + 256] + b[tid + 256]);
  op[tid + 512] = (_Float16)(d2 * rstd * g[tid + 512] + b[tid + 512]);
}

// ---------------- WMMA GEMM: C[M,N] = A[M,K](f16) * Bt[N,K](f16, pre-transposed) --------
// Block tile 128x128, BK=32. 256 thr = 8 waves in a 4x2 grid; each wave owns a
// 32x64 strip -> 8 WMMAs per k-step, B fragments reused across two A fragments.
// Both LDS tiles are staged with contiguous b128 copies; on CDNA5 these go
// through GLOBAL_LOAD_ASYNC_TO_LDS_B128 (ASYNCcnt), no VGPR round trip.
// Epilogue: (+bias)*scale, optional QuickGELU (v_rcp), optional f32 residual,
// stores f32 and/or f16. M % 128 == 0, N % 128 == 0, K % 32 == 0 (true here).
__global__ __launch_bounds__(256)
void gemm_wmma_f16(const _Float16* __restrict__ A, const _Float16* __restrict__ Bt,
                   const float* __restrict__ bias, const float* __restrict__ resid,
                   float* __restrict__ outF, _Float16* __restrict__ outH,
                   int Ndim, int Kdim, float scale, int gelu)
{
  __shared__ _Float16 As[128][32];   // 8 KB, [m][k]
  __shared__ _Float16 Bs[128][32];   // 8 KB, [n][k]
  int tid = threadIdx.x;
  int wv = tid >> 5, lane = tid & 31;
  int row16 = lane & 15, hi = lane >> 4;
  int bm = blockIdx.y * 128, bn = blockIdx.x * 128;
  int wr = (wv >> 1) * 32;           // wave row offset: 0,32,64,96
  int wc = (wv & 1) * 64;            // wave col offset: 0,64
  int sr = tid >> 2, sc = (tid & 3) << 3;   // staging: row 0..63 (+64), 16B chunk

  v8f acc[2][4] = {};
  for (int k0 = 0; k0 < Kdim; k0 += 32) {
    const _Float16* ag0 = A  + (size_t)(bm + sr) * Kdim + k0 + sc;
    const _Float16* ag1 = ag0 + (size_t)64 * Kdim;
    const _Float16* bg0 = Bt + (size_t)(bn + sr) * Kdim + k0 + sc;
    const _Float16* bg1 = bg0 + (size_t)64 * Kdim;
#if USE_ASYNC_LDS
    __builtin_amdgcn_global_load_async_to_lds_b128((gas_i4*)ag0, (las_i4*)&As[sr][sc], 0, 0);
    __builtin_amdgcn_global_load_async_to_lds_b128((gas_i4*)ag1, (las_i4*)&As[sr + 64][sc], 0, 0);
    __builtin_amdgcn_global_load_async_to_lds_b128((gas_i4*)bg0, (las_i4*)&Bs[sr][sc], 0, 0);
    __builtin_amdgcn_global_load_async_to_lds_b128((gas_i4*)bg1, (las_i4*)&Bs[sr + 64][sc], 0, 0);
#else
    float4 ra0 = *(const float4*)ag0;
    float4 ra1 = *(const float4*)ag1;
    float4 rb0 = *(const float4*)bg0;
    float4 rb1 = *(const float4*)bg1;
    *(float4*)&As[sr][sc]      = ra0;
    *(float4*)&As[sr + 64][sc] = ra1;
    *(float4*)&Bs[sr][sc]      = rb0;
    *(float4*)&Bs[sr + 64][sc] = rb1;
#endif
    // Prefetch next k-slice while this tile is consumed (global_prefetch_b8)
    if (k0 + 32 < Kdim) {
      __builtin_prefetch(ag0 + 32, 0, 1);
      __builtin_prefetch(bg0 + 32, 0, 1);
    }
#if USE_ASYNC_LDS
    wait_asynccnt0();
#endif
    __syncthreads();

    // A fragments (16x32): lane row = lane&15, elems 0..7 -> K=hi*8.., 8..15 -> K=16+hi*8..
    V16 a0, a1;
    int kb = hi << 3;
    a0.f[0] = *(const float4*)&As[wr + row16][kb];
    a0.f[1] = *(const float4*)&As[wr + row16][16 + kb];
    a1.f[0] = *(const float4*)&As[wr + 16 + row16][kb];
    a1.f[1] = *(const float4*)&As[wr + 16 + row16][16 + kb];
#pragma unroll
    for (int nt = 0; nt < 4; ++nt) {
      // B fragment (32x16): lane col = lane&15, elems 0..7 -> K=hi*8.., 8..15 -> K=16+hi*8..
      V16 bf;
      int nl = wc + nt * 16 + row16, kh = hi << 3;
      bf.f[0] = *(const float4*)&Bs[nl][kh];
      bf.f[1] = *(const float4*)&Bs[nl][kh + 16];
      acc[0][nt] = __builtin_amdgcn_wmma_f32_16x16x32_f16(false, a0.v, false, bf.v,
                                                          (short)0, acc[0][nt], false, false);
      acc[1][nt] = __builtin_amdgcn_wmma_f32_16x16x32_f16(false, a1.v, false, bf.v,
                                                          (short)0, acc[1][nt], false, false);
    }
    __syncthreads();
  }

  // Epilogue. C layout: VGPR r -> row = r + hi*8; col = lane&15
#pragma unroll
  for (int i = 0; i < 2; ++i)
#pragma unroll
    for (int nt = 0; nt < 4; ++nt) {
      int col = bn + wc + nt * 16 + row16;
      float bv = bias[col];
#pragma unroll
      for (int r = 0; r < 8; ++r) {
        int rowg = bm + wr + i * 16 + r + hi * 8;
        float c = (acc[i][nt][r] + bv) * scale;
        if (gelu) c = c * __builtin_amdgcn_rcpf(1.f + __expf(-1.702f * c));  // QuickGELU
        if (resid) c += resid[(size_t)rowg * Ndim + col];
        if (outF) outF[(size_t)rowg * Ndim + col] = c;
        if (outH) outH[(size_t)rowg * Ndim + col] = (_Float16)c;
      }
    }
}

// ---------------- V transpose: [T,768] f16 -> vT[nh*64+d][224] (zero-padded) ----------------
__global__ __launch_bounds__(256)
void transpose_v_kernel(const _Float16* __restrict__ v, _Float16* __restrict__ vT) {
  int nh = blockIdx.x;                 // 0..128*12-1
  int n = nh / NHEAD, h = nh % NHEAD;
  for (int idx = threadIdx.x; idx < HD * 224; idx += 256) {
    int d = idx / 224, l = idx % 224;
    _Float16 val = (_Float16)0.f;
    if (l < LSEQ) val = v[((size_t)(n * LSEQ + l)) * CDIM + h * HD + d];
    vT[((size_t)nh * HD + d) * 224 + l] = val;
  }
}

// ---------------- Flash attention with WMMA ----------------
// Block = 128 thr (4 waves). Each wave handles 16 queries of one (n,h).
// grid.x = 128*12*4 ; qtile = (blockIdx&3)*4 + wave ; 13 tiles cover 197 (pad to 208).
__global__ __launch_bounds__(128)
void attn_wmma_kernel(const _Float16* __restrict__ q, const _Float16* __restrict__ k,
                      const _Float16* __restrict__ vT, _Float16* __restrict__ mix)
{
  __shared__ _Float16 Pst[4][16][32];  // per-wave P staging (C-layout -> A-layout)
  int wv = threadIdx.x >> 5, lane = threadIdx.x & 31;
  int col = lane & 15, hi = lane >> 4;
  int nh = blockIdx.x >> 2;
  int n = nh / NHEAD, h = nh % NHEAD;
  int qt = (blockIdx.x & 3) * 4 + wv;
  if (qt >= 13) return;                // wave-uniform exit

  // Q fragments (A, 16x32) for d=[0,32) and [32,64). Q is pre-scaled by 1/8 in its GEMM.
  int qrow = qt * 16 + col; if (qrow > LSEQ - 1) qrow = LSEQ - 1;
  const _Float16* qp = q + ((size_t)(n * LSEQ + qrow)) * CDIM + h * HD;
  int kb = hi << 3;
  V16 qa0, qa1;
  qa0.f[0] = *(const float4*)(qp + kb);       qa0.f[1] = *(const float4*)(qp + 16 + kb);
  qa1.f[0] = *(const float4*)(qp + 32 + kb);  qa1.f[1] = *(const float4*)(qp + 48 + kb);

  float m[8], l[8], aset[8];
#pragma unroll
  for (int r = 0; r < 8; ++r) { m[r] = -1e30f; l[r] = 0.f; }
  v8f O[4] = {};

  for (int kt = 0; kt < 7; ++kt) {     // 32 keys per iteration (224 >= 197)
    int keyA = kt * 32 + col;
    int keyB = keyA + 16;
    int ca = keyA > LSEQ - 1 ? LSEQ - 1 : keyA;
    int cb = keyB > LSEQ - 1 ? LSEQ - 1 : keyB;
    const _Float16* kpA = k + ((size_t)(n * LSEQ + ca)) * CDIM + h * HD;
    const _Float16* kpB = k + ((size_t)(n * LSEQ + cb)) * CDIM + h * HD;
    int kh16 = hi << 3;
    // Score B fragments (K^T 32x16): col = key, elems 0..7 -> d = dstep*32 + hi*8.., 8..15 -> d = dstep*32 + 16 + hi*8..
    V16 fA0, fA1, fB0, fB1;
    fA0.f[0] = *(const float4*)(kpA + kh16);        fA0.f[1] = *(const float4*)(kpA + kh16 + 16);
    fA1.f[0] = *(const float4*)(kpA + 32 + kh16);   fA1.f[1] = *(const float4*)(kpA + 32 + kh16 + 16);
    fB0.f[0] = *(const float4*)(kpB + kh16);        fB0.f[1] = *(const float4*)(kpB + kh16 + 16);
    fB1.f[0] = *(const float4*)(kpB + 32 + kh16);   fB1.f[1] = *(const float4*)(kpB + 32 + kh16 + 16);

    v8f sa = {}, sb = {};
    sa = __builtin_amdgcn_wmma_f32_16x16x32_f16(false, qa0.v, false, fA0.v, (short)0, sa, false, false);
    sa = __builtin_amdgcn_wmma_f32_16x16x32_f16(false, qa1.v, false, fA1.v, (short)0, sa, false, false);
    sb = __builtin_amdgcn_wmma_f32_16x16x32_f16(false, qa0.v, false, fB0.v, (short)0, sb, false, false);
    sb = __builtin_amdgcn_wmma_f32_16x16x32_f16(false, qa1.v, false, fB1.v, (short)0, sb, false, false);

    bool okA = (keyA < LSEQ), okB = (keyB < LSEQ);
    // Online softmax: stats per row, butterflied within each 16-lane half
#pragma unroll
    for (int r = 0; r < 8; ++r) {
      float sva = okA ? sa[r] : -1e30f;
      float svb = okB ? sb[r] : -1e30f;
      float mx = fmaxf(sva, svb);
      mx = fmaxf(mx, __shfl_xor(mx, 1));
      mx = fmaxf(mx, __shfl_xor(mx, 2));
      mx = fmaxf(mx, __shfl_xor(mx, 4));
      mx = fmaxf(mx, __shfl_xor(mx, 8));
      float nm = fmaxf(m[r], mx);
      float alpha = __expf(m[r] - nm);
      float pa = __expf(sva - nm);
      float pb = __expf(svb - nm);
      float ts = pa + pb;
      ts += __shfl_xor(ts, 1); ts += __shfl_xor(ts, 2);
      ts += __shfl_xor(ts, 4); ts += __shfl_xor(ts, 8);
      l[r] = l[r] * alpha + ts;
      m[r] = nm; aset[r] = alpha;
      Pst[wv][r + hi * 8][col]      = (_Float16)pa;   // C-layout -> LDS
      Pst[wv][r + hi * 8][col + 16] = (_Float16)pb;
    }
#pragma unroll
    for (int nt = 0; nt < 4; ++nt)
#pragma unroll
      for (int r = 0; r < 8; ++r) O[nt][r] *= aset[r];

    // Per-wave LDS is processed in order; wait + fence before A-layout reload
    asm volatile("s_wait_dscnt 0x0" ::: "memory");

    V16 pA;  // P as A fragment 16x32 over the key dim
    pA.f[0] = *(const float4*)&Pst[wv][col][kb];
    pA.f[1] = *(const float4*)&Pst[wv][col][16 + kb];
#pragma unroll
    for (int nt = 0; nt < 4; ++nt) {
      // V^T B fragment: col = d, elems 0..7 -> key = kt*32 + hi*8.., 8..15 -> key = kt*32 + 16 + hi*8..
      const _Float16* vp = vT + ((size_t)nh * HD + nt * 16 + col) * 224 + kt * 32 + (hi << 3);
      V16 vB;
      vB.f[0] = *(const float4*)vp;
      vB.f[1] = *(const float4*)(vp + 16);
      O[nt] = __builtin_amdgcn_wmma_f32_16x16x32_f16(false, pA.v, false, vB.v,
                                                     (short)0, O[nt], false, false);
    }
  }

  // Normalize (v_rcp) and store mix (f16) — rows follow the C layout
#pragma unroll
  for (int r = 0; r < 8; ++r) aset[r] = __builtin_amdgcn_rcpf(l[r]);
#pragma unroll
  for (int nt = 0; nt < 4; ++nt)
#pragma unroll
    for (int r = 0; r < 8; ++r) {
      int ql = qt * 16 + r + hi * 8;
      if (ql < LSEQ) {
        float val = O[nt][r] * aset[r];
        mix[((size_t)(n * LSEQ + ql)) * CDIM + h * HD + nt * 16 + col] = (_Float16)val;
      }
    }
}

// ---------------- host orchestration ----------------
extern "C" void kernel_launch(void* const* d_in, const int* in_sizes, int n_in,
                              void* d_out, int out_size, void* d_ws, size_t ws_size,
                              hipStream_t stream) {
  (void)in_sizes; (void)n_in; (void)out_size; (void)ws_size;
  const float* x   = (const float*)d_in[0];
  const float* Wq  = (const float*)d_in[1];
  const float* bq  = (const float*)d_in[2];
  const float* Wk  = (const float*)d_in[3];
  const float* bk  = (const float*)d_in[4];
  const float* Wv  = (const float*)d_in[5];
  const float* bv  = (const float*)d_in[6];
  const float* Wo  = (const float*)d_in[7];
  const float* bo  = (const float*)d_in[8];
  const float* W1  = (const float*)d_in[9];
  const float* b1  = (const float*)d_in[10];
  const float* W2  = (const float*)d_in[11];
  const float* b2  = (const float*)d_in[12];
  const float* g1  = (const float*)d_in[13];
  const float* be1 = (const float*)d_in[14];
  const float* g2  = (const float*)d_in[15];
  const float* be2 = (const float*)d_in[16];
  float* out = (float*)d_out;

  char* ws = (char*)d_ws;
  size_t off = 0;
  auto alloc = [&](size_t bytes) { size_t o = off; off += (bytes + 255) & ~(size_t)255; return o; };

  const size_t W768 = (size_t)768 * 768;       // 589824
  const size_t WBIG = (size_t)768 * 3072;      // 2359296
  const size_t TC   = (size_t)TOK * CDIM;
  const size_t TM   = (size_t)TOK * MDIM;
  const size_t VTN  = (size_t)128 * NHEAD * HD * 224;

  _Float16* wqh = (_Float16*)(ws + alloc(W768 * 2));   // all weights stored [N][K] f16
  _Float16* wkh = (_Float16*)(ws + alloc(W768 * 2));
  _Float16* wvh = (_Float16*)(ws + alloc(W768 * 2));
  _Float16* woh = (_Float16*)(ws + alloc(W768 * 2));
  _Float16* w1h = (_Float16*)(ws + alloc(WBIG * 2));
  _Float16* w2h = (_Float16*)(ws + alloc(WBIG * 2));
  _Float16* xnh = (_Float16*)(ws + alloc(TC * 2));   // LN out; reused for attn mix, LN2 out
  float*    x1f = (float*)   (ws + alloc(TC * 4));   // residual stream after attention
  _Float16* qh  = (_Float16*)(ws + alloc(TC * 2));
  _Float16* khb = (_Float16*)(ws + alloc(TC * 2));
  _Float16* vhb = (_Float16*)(ws + alloc(TC * 2));
  _Float16* vTh = (_Float16*)(ws + alloc(VTN * 2));
  _Float16* hb  = qh;  // MLP hidden (TM*2 = 155 MB) overlays dead q/k/v/vT region (160 MB)
  (void)TM;

  // Weight conversions f32[K][N] -> f16[N][K]
  int g768 = (int)((W768 + 255) / 256), gbig = (int)((WBIG + 255) / 256);
  cvtT_f32_f16_kernel<<<g768, 256, 0, stream>>>(Wq, wqh, CDIM, CDIM);
  cvtT_f32_f16_kernel<<<g768, 256, 0, stream>>>(Wk, wkh, CDIM, CDIM);
  cvtT_f32_f16_kernel<<<g768, 256, 0, stream>>>(Wv, wvh, CDIM, CDIM);
  cvtT_f32_f16_kernel<<<g768, 256, 0, stream>>>(Wo, woh, CDIM, CDIM);
  cvtT_f32_f16_kernel<<<gbig, 256, 0, stream>>>(W1, w1h, CDIM, MDIM);
  cvtT_f32_f16_kernel<<<gbig, 256, 0, stream>>>(W2, w2h, MDIM, CDIM);

  // LN1
  layernorm_f16_kernel<<<TOK, 256, 0, stream>>>(x, g1, be1, xnh);

  // QKV projections (scale 1/sqrt(64) folded into Q)
  dim3 gqkv(CDIM / 128, TOK / 128);
  gemm_wmma_f16<<<gqkv, 256, 0, stream>>>(xnh, wqh, bq, nullptr, nullptr, qh,  CDIM, CDIM, 0.125f, 0);
  gemm_wmma_f16<<<gqkv, 256, 0, stream>>>(xnh, wkh, bk, nullptr, nullptr, khb, CDIM, CDIM, 1.f,    0);
  gemm_wmma_f16<<<gqkv, 256, 0, stream>>>(xnh, wvh, bv, nullptr, nullptr, vhb, CDIM, CDIM, 1.f,    0);

  // V^T staging + flash attention (mix reuses xnh)
  transpose_v_kernel<<<128 * NHEAD, 256, 0, stream>>>(vhb, vTh);
  attn_wmma_kernel<<<128 * NHEAD * 4, 128, 0, stream>>>(qh, khb, vTh, xnh);

  // Output projection + residual(x) -> x1 (f32)
  gemm_wmma_f16<<<gqkv, 256, 0, stream>>>(xnh, woh, bo, x, x1f, nullptr, CDIM, CDIM, 1.f, 0);

  // LN2
  layernorm_f16_kernel<<<TOK, 256, 0, stream>>>(x1f, g2, be2, xnh);

  // MLP: fc1 + QuickGELU -> f16 hidden; fc2 + residual(x1) -> d_out (f32)
  dim3 gfc1(MDIM / 128, TOK / 128);
  gemm_wmma_f16<<<gfc1, 256, 0, stream>>>(xnh, w1h, b1, nullptr, nullptr, hb, MDIM, CDIM, 1.f, 1);
  dim3 gfc2(CDIM / 128, TOK / 128);
  gemm_wmma_f16<<<gfc2, 256, 0, stream>>>(hb, w2h, b2, x1f, out, nullptr, CDIM, MDIM, 1.f, 0);
}
